// QLoRABigNet_6416681140836
// MI455X (gfx1250) — matrix operand order repaired and checked by reading, not compiled, not run on hardware
//
#include <hip/hip_runtime.h>

typedef __attribute__((ext_vector_type(16))) _Float16 v16h;
typedef __attribute__((ext_vector_type(8)))  _Float16 v8h;
typedef __attribute__((ext_vector_type(4)))  _Float16 v4h;
typedef __attribute__((ext_vector_type(8)))  float    v8f;

#define D_DIM  1024
#define R_DIM  32
#define QGROUP 16
#define NLAYER 18

#define BM 128
#define BN 256
#define BK 32
#define LDA 40   // padded LDS row stride in halves (80 B, keeps 16B-aligned chunks)

// ---------------------------------------------------------------------------
// Prep: W_eff[l][o][i] = f16( qw*scale + sum_r lb[o][r]*la[r][i] )
// Folds the rank-32 LoRA into the dequantized weight once per call.
// ---------------------------------------------------------------------------
__global__ __launch_bounds__(256)
void prep_weff(const int* __restrict__ qw, const float* __restrict__ sc,
               const float* __restrict__ la, const float* __restrict__ lb,
               _Float16* __restrict__ weff) {
  const int l   = blockIdx.y;
  const int e   = blockIdx.x * blockDim.x + threadIdx.x;  // one thread per 4 elems
  const int idx = e * 4;                                  // flat o*1024 + i
  const int o   = idx >> 10;
  const int i0  = idx & (D_DIM - 1);

  int4  q = *(const int4*)(qw + (size_t)l * D_DIM * D_DIM + idx);
  float s = sc[(size_t)l * D_DIM * (D_DIM / QGROUP) + o * (D_DIM / QGROUP) + (i0 >> 4)];
  float a0 = q.x * s, a1 = q.y * s, a2 = q.z * s, a3 = q.w * s;

  const float* lap = la + (size_t)l * R_DIM * D_DIM + i0;
  const float* lbp = lb + (size_t)l * D_DIM * R_DIM + o * R_DIM;
  for (int r = 0; r < R_DIM; ++r) {
    float4 a4 = *(const float4*)(lap + r * D_DIM);
    float  b  = lbp[r];
    a0 += b * a4.x; a1 += b * a4.y; a2 += b * a4.z; a3 += b * a4.w;
  }
  v4h h4 = { (_Float16)a0, (_Float16)a1, (_Float16)a2, (_Float16)a3 };
  *(v4h*)(weff + (size_t)l * D_DIM * D_DIM + idx) = h4;
}

// ---------------------------------------------------------------------------
// GEMM: C[n][o] = relu?( sum_i A[n][i] * W[o][i] + bias[o] )
// 128x256 block tile, 256 threads (8 waves, 2x4), 64x64 per wave.
// f16 WMMA with fp32 accumulate; double-buffered LDS, 1 barrier per K-step.
// ---------------------------------------------------------------------------
__global__ __launch_bounds__(256)
void gemm_qlora(const float* __restrict__ A, const _Float16* __restrict__ W,
                const float* __restrict__ bias, float* __restrict__ C, int relu) {
  __shared__ _Float16 sA[2][BM * LDA];   // 2 x 10.0 KB
  __shared__ _Float16 sB[2][BN * LDA];   // 2 x 20.0 KB

  const int tid    = threadIdx.x;
  const int lane   = tid & 31;
  const int wave   = tid >> 5;      // 0..7
  const int waveM  = wave & 1;      // 2 waves along M
  const int waveN  = wave >> 1;     // 4 waves along N
  const int laneLo = lane & 15;
  const int laneHi = lane >> 4;     // half-wave select

  const int rowBase = blockIdx.y * BM;
  const int colBase = blockIdx.x * BN;

  v8f acc[4][4] = {};               // 64x64 per wave = 4x4 fragments

  // ---- tile staging (global -> LDS, A converted fp32->f16) ----
  auto stage = [&](int bi, int kb) {
    #pragma unroll
    for (int it = 0; it < 4; ++it) {            // A: 128x32 fp32 = 1024 float4
      int id = tid + it * 256;
      int r  = id >> 3;                         // 8 float4 per row
      int c4 = id & 7;
      float4 a4 = *(const float4*)(A + (size_t)(rowBase + r) * D_DIM + kb * BK + c4 * 4);
      v4h h4 = { (_Float16)a4.x, (_Float16)a4.y, (_Float16)a4.z, (_Float16)a4.w };
      *(v4h*)(&sA[bi][r * LDA + c4 * 4]) = h4;
    }
    #pragma unroll
    for (int it = 0; it < 4; ++it) {            // B: 256x32 f16 = 1024 v8h
      int id = tid + it * 256;
      int r  = id >> 2;                         // 4 v8h per row
      int c8 = id & 3;
      v8h w8 = *(const v8h*)(W + (size_t)(colBase + r) * D_DIM + kb * BK + c8 * 8);
      *(v8h*)(&sB[bi][r * LDA + c8 * 8]) = w8;
    }
  };

  stage(0, 0);
  __syncthreads();

  #pragma unroll 2
  for (int kb = 0; kb < D_DIM / BK; ++kb) {
    const int cur = kb & 1;
    if (kb + 1 < D_DIM / BK) stage(cur ^ 1, kb + 1);   // prefetch next tile

    // --- fragment loads per documented CDNA5 16-bit A/B VGPR layouts ---
    union Frag { v16h f; v8h h[2]; };
    Frag a[4], b[4];
    #pragma unroll
    for (int mi = 0; mi < 4; ++mi) {
      int row = waveM * 64 + mi * 16 + laneLo;         // M = laneLo
      // e=0..7 -> K = 8*laneHi + 0..7 ; e=8..15 -> K = 16 + 8*laneHi + 0..7
      a[mi].h[0] = *(const v8h*)(&sA[cur][row * LDA + laneHi * 8]);
      a[mi].h[1] = *(const v8h*)(&sA[cur][row * LDA + 16 + laneHi * 8]);
    }
    #pragma unroll
    for (int ni = 0; ni < 4; ++ni) {
      int row = waveN * 64 + ni * 16 + laneLo;         // N = laneLo
      // e=0..7 -> K = 8*laneHi + 0..7 ; e=8..15 -> K = 16 + 8*laneHi + 0..7
      b[ni].h[0] = *(const v8h*)(&sB[cur][row * LDA + laneHi * 8]);
      b[ni].h[1] = *(const v8h*)(&sB[cur][row * LDA + 16 + laneHi * 8]);
    }

    #pragma unroll
    for (int mi = 0; mi < 4; ++mi)
      #pragma unroll
      for (int ni = 0; ni < 4; ++ni)
        acc[mi][ni] = __builtin_amdgcn_wmma_f32_16x16x32_f16(
            false, a[mi].f, false, b[ni].f, (short)0, acc[mi][ni], false, false);

    __syncthreads();   // next tile staged + current reads done
  }

  // --- epilogue: bias (+ optional ReLU), fp32 store ---
  #pragma unroll
  for (int ni = 0; ni < 4; ++ni) {
    int col = colBase + waveN * 64 + ni * 16 + laneLo;
    float bv = bias[col];
    #pragma unroll
    for (int mi = 0; mi < 4; ++mi) {
      #pragma unroll
      for (int r = 0; r < 8; ++r) {
        int row = rowBase + waveM * 64 + mi * 16 + r + laneHi * 8;
        float v = acc[mi][ni][r] + bv;
        if (relu) v = fmaxf(v, 0.f);
        C[(size_t)row * D_DIM + col] = v;
      }
    }
  }
}

// ---------------------------------------------------------------------------
// h = LayerNorm(y + h) * g + b      (one 256-thread block per row)
// ---------------------------------------------------------------------------
__global__ __launch_bounds__(256)
void residual_ln(const float* __restrict__ y, const float* __restrict__ hin,
                 const float* __restrict__ g, const float* __restrict__ bta,
                 float* __restrict__ hout) {
  __shared__ float red0[256];
  __shared__ float red1[256];
  const int row = blockIdx.x;
  const int tid = threadIdx.x;
  const float* yp = y   + (size_t)row * D_DIM;
  const float* hp = hin + (size_t)row * D_DIM;

  float vals[4];
  float s = 0.f, s2 = 0.f;
  #pragma unroll
  for (int i = 0; i < 4; ++i) {
    int c = tid + i * 256;
    float v = yp[c] + hp[c];
    vals[i] = v;
    s += v; s2 += v * v;
  }
  red0[tid] = s; red1[tid] = s2;
  __syncthreads();
  for (int off = 128; off > 0; off >>= 1) {
    if (tid < off) { red0[tid] += red0[tid + off]; red1[tid] += red1[tid + off]; }
    __syncthreads();
  }
  float mean = red0[0] * (1.f / D_DIM);
  float var  = red1[0] * (1.f / D_DIM) - mean * mean;
  float inv  = rsqrtf(var + 1e-5f);
  #pragma unroll
  for (int i = 0; i < 4; ++i) {
    int c = tid + i * 256;
    hout[(size_t)row * D_DIM + c] = (vals[i] - mean) * inv * g[c] + bta[c];
  }
}

// ---------------------------------------------------------------------------
// y += h  (final block: no layernorm; y aliases d_out, elementwise-safe)
// ---------------------------------------------------------------------------
__global__ __launch_bounds__(256)
void residual_add(const float* __restrict__ hin, float* __restrict__ y) {
  size_t id = ((size_t)blockIdx.x * blockDim.x + threadIdx.x) * 4;
  float4 a = *(const float4*)(hin + id);
  float4 b = *(float4*)(y + id);
  b.x += a.x; b.y += a.y; b.z += a.z; b.w += a.w;
  *(float4*)(y + id) = b;
}

// ---------------------------------------------------------------------------
extern "C" void kernel_launch(void* const* d_in, const int* in_sizes, int n_in,
                              void* d_out, int out_size, void* d_ws, size_t ws_size,
                              hipStream_t stream) {
  const float* x    = (const float*)d_in[0];
  const int*   qw   = (const int*)  d_in[1];
  const float* sc   = (const float*)d_in[2];
  const float* bias = (const float*)d_in[3];
  const float* la   = (const float*)d_in[4];
  const float* lb   = (const float*)d_in[5];
  const float* lng  = (const float*)d_in[6];
  const float* lnb  = (const float*)d_in[7];
  const int N = in_sizes[0] / D_DIM;   // 16384

  // workspace layout: [W_eff f16 | h fp32 | bufB fp32]; bufA = d_out
  char*      ws    = (char*)d_ws;
  _Float16*  weff  = (_Float16*)ws;
  size_t     wB    = (size_t)NLAYER * D_DIM * D_DIM * sizeof(_Float16);
  size_t     actB  = (size_t)N * D_DIM * sizeof(float);
  float*     h     = (float*)(ws + wB);
  float*     bufB  = (float*)(ws + wB + actB);
  float*     bufA  = (float*)d_out;

  // 1) fold dequant + LoRA into f16 effective weights (once per call)
  prep_weff<<<dim3(D_DIM * D_DIM / (4 * 256), NLAYER), 256, 0, stream>>>(qw, sc, la, lb, weff);

  // 2) residual stream starts at x
  hipMemcpyAsync(h, x, actB, hipMemcpyDeviceToDevice, stream);

  dim3 ggrid(D_DIM / BN, N / BM);
  for (int blk = 0; blk < 6; ++blk) {
    int l0 = blk * 3;
    const _Float16* w0 = weff + (size_t)(l0 + 0) * D_DIM * D_DIM;
    const _Float16* w1 = weff + (size_t)(l0 + 1) * D_DIM * D_DIM;
    const _Float16* w2 = weff + (size_t)(l0 + 2) * D_DIM * D_DIM;

    gemm_qlora<<<ggrid, 256, 0, stream>>>(h,    w0, bias + (l0 + 0) * D_DIM, bufA, 1);
    gemm_qlora<<<ggrid, 256, 0, stream>>>(bufA, w1, bias + (l0 + 1) * D_DIM, bufB, 1);
    gemm_qlora<<<ggrid, 256, 0, stream>>>(bufB, w2, bias + (l0 + 2) * D_DIM, bufA, 0);

    if (blk < 5) {
      residual_ln<<<N, 256, 0, stream>>>(bufA, h, lng + blk * D_DIM, lnb + blk * D_DIM, h);
    } else {
      residual_add<<<(N * D_DIM) / (256 * 4), 256, 0, stream>>>(h, bufA);  // bufA == d_out
    }
  }
}
